// SegTransformerDecoder_16106127360659
// MI455X (gfx1250) — matrix operand order repaired and checked by reading, not compiled, not run on hardware
//
#include <hip/hip_runtime.h>
#include <hip/hip_bf16.h>
#include <cstdint>
#include <cstddef>

// ---------------------------------------------------------------------------
// MI455X (gfx1250) implementation.
//  B=1, E=256, H=W=200 (Q=40000 tokens), P=4, M=4, F=128, DH=32.
//  ~161 GFLOP vs ~273 MB HBM traffic -> f16 WMMA (f32 accum) for every GEMM
//  so compute time matches the ~12us memory floor at 23.3 TB/s.
//  MLP GEMMs double-buffer K-chunks through LDS using the CDNA5 async
//  global->LDS path (GLOBAL_LOAD_ASYNC_TO_LDS_B128 + ASYNCcnt), emitted via
//  inline asm because the toolchain's builtin uses an unspellable LangAS.
// ---------------------------------------------------------------------------

typedef __attribute__((ext_vector_type(16))) _Float16 v16h;
typedef __attribute__((ext_vector_type(8)))  _Float16 v8h;
typedef __attribute__((ext_vector_type(8)))  float    v8f;

#define Q_TOKENS 40000
#define F_DIM    128
#define E_DIM    256

// ---- CDNA5 async global->LDS (inline asm; device pass only) ---------------
#if defined(__HIP_DEVICE_COMPILE__)
#define HAVE_ASYNC_LDS 1
#else
#define HAVE_ASYNC_LDS 0
#endif

#if HAVE_ASYNC_LDS
// One 16-byte async copy, tracked with ASYNCcnt (no VGPR data movement).
// VDST = LDS byte address (low 32 bits of the generic LDS pointer),
// VADDR = 64-bit global address, no SADDR ("off").
static __device__ __forceinline__ void async_cp16(const _Float16* g, _Float16* l) {
  asm volatile("global_load_async_to_lds_b128 %0, %1, off"
               :
               : "v"((unsigned)(uintptr_t)l),
                 "v"((unsigned long long)(uintptr_t)g)
               : "memory");
}
static __device__ __forceinline__ void async_wait3() {
  asm volatile("s_wait_asynccnt 0x3" ::: "memory");
}
static __device__ __forceinline__ void async_wait0() {
  asm volatile("s_wait_asynccnt 0x0" ::: "memory");
}
#endif

// ---- WMMA fragment helpers (layouts per CDNA5 ISA 7.12.2, wave32) ----------

// A fragment, 16x32 f16. Row M = lane%16. halves 0..7 -> K = 8*(lane/16)+h,
// halves 8..15 -> K = 16 + 8*(lane/16)+h. Two contiguous 16B loads.
static __device__ __forceinline__ v16h wmma_ld_a(const _Float16* base, int stride, int lane) {
  const int m  = lane & 15;
  const int hi = lane >> 4;
  const _Float16* p = base + (size_t)m * stride + hi * 8;
  v8h lo = *(const v8h*)(p);
  v8h hh = *(const v8h*)(p + 16);
  return __builtin_shufflevector(lo, hh, 0,1,2,3,4,5,6,7,8,9,10,11,12,13,14,15);
}

// B fragment, 32x16 f16, sourced from an [N][K] (row-major, "W[out][in]")
// array: col N = lane%16, halves 0..7 -> K = 8*(lane/16)+h,
// halves 8..15 -> K = 16 + 8*(lane/16)+h.
static __device__ __forceinline__ v16h wmma_ld_b(const _Float16* base, int stride, int lane) {
  const int n  = lane & 15;
  const int hi = lane >> 4;
  const _Float16* p = base + (size_t)n * stride + hi * 8;
  v8h lo = *(const v8h*)(p);
  v8h hh = *(const v8h*)(p + 16);
  return __builtin_shufflevector(lo, hh, 0,1,2,3,4,5,6,7,8,9,10,11,12,13,14,15);
}

static __device__ __forceinline__ v8f wmma_f16(v16h a, v16h b, v8f c) {
  // v_wmma_f32_16x16x32_f16 (8-arg: neg_a, A, neg_b, B, c_mod, C, reuse_a, reuse_b)
  return __builtin_amdgcn_wmma_f32_16x16x32_f16(false, a, false, b, (short)0, c, false, false);
}

static __device__ __forceinline__ float gelu_erf(float x) {
  return 0.5f * x * (1.0f + erff(x * 0.70710678118654752440f));
}

// ---------------------------------------------------------------------------
// f32 -> f16 weight conversion
// ---------------------------------------------------------------------------
__global__ void cvt_f32_to_f16_kernel(const float* __restrict__ src,
                                      _Float16* __restrict__ dst, int n) {
  int i = blockIdx.x * 256 + threadIdx.x;
  if (i < n) dst[i] = (_Float16)src[i];
}

// ---------------------------------------------------------------------------
// Fused front-end: q_proj -> to_q, k_proj -> to_k, P=4 attention, weighted
// features.  One workgroup (8 waves) handles 16 tokens.  All four projections
// run through v_wmma_f32_16x16x32_f16 with operands staged in LDS.
// ---------------------------------------------------------------------------
__global__ __launch_bounds__(256)
void frontend_attn_kernel(const float* __restrict__ bev_query,     // [256, 40000]
                          const float* __restrict__ sampled_feat,  // [40000*512]
                          const float* __restrict__ bev_pos_embed, // [40000*128]
                          const float* __restrict__ pos_embed_2d,  // [40000*512]
                          const _Float16* __restrict__ wq,   const float* __restrict__ bq,
                          const _Float16* __restrict__ wtq,  const float* __restrict__ btq,
                          const _Float16* __restrict__ wk,   const float* __restrict__ bk,
                          const _Float16* __restrict__ wtk,  const float* __restrict__ btk,
                          _Float16* __restrict__ xout)             // [40000, 512] f16
{
  __shared__ _Float16 sQ [16 * 256];   // f16 query tile
  __shared__ _Float16 sQA[16 * 128];   // q_proj(query)+pe
  __shared__ _Float16 sQB[16 * 128];   // to_q(...)
  __shared__ _Float16 sK [64 * 128];   // key tile, later to_k result
  __shared__ _Float16 sK2[64 * 128];   // k_proj(key)+pe
  __shared__ float    sAtt[16 * 16];   // softmax weights [tl][m][p]

  const int tid  = threadIdx.x;
  const int lane = tid & 31;
  const int wave = tid >> 5;
  const int hi   = lane >> 4;
  const int t0   = blockIdx.x * 16;

  // ---- stage query (transposed gather) and key (contiguous) ----
  for (int i = tid; i < 16 * 256; i += 256) {
    int tl = i & 15;
    int e  = i >> 4;
    sQ[tl * 256 + e] = (_Float16)bev_query[(size_t)e * Q_TOKENS + t0 + tl];
  }
  {
    const float4* kp = (const float4*)(sampled_feat + (size_t)t0 * 512);
    for (int i = 0; i < 8; ++i) {
      int v = tid + i * 256;          // 0..2047 float4s == 64x128 floats
      float4 d = kp[v];
      int e = v * 4;
      sK[e + 0] = (_Float16)d.x;
      sK[e + 1] = (_Float16)d.y;
      sK[e + 2] = (_Float16)d.z;
      sK[e + 3] = (_Float16)d.w;
    }
  }
  __syncthreads();

  // ---- q_proj: (16x256)@(256->128) + bias + bev_pos_embed -> sQA ----
  {
    const int n0 = wave * 16;
    v8f acc = {};
    for (int kc = 0; kc < 256; kc += 32) {
      v16h a = wmma_ld_a(sQ + kc, 256, lane);
      v16h b = wmma_ld_b(wq + (size_t)n0 * 256 + kc, 256, lane);
      acc = wmma_f16(a, b, acc);
    }
    const int n = n0 + (lane & 15);
    const float bias = bq[n];
    for (int r = 0; r < 8; ++r) {
      int m = r + hi * 8;
      float pe = bev_pos_embed[(size_t)(t0 + m) * 128 + n];
      sQA[m * 128 + n] = (_Float16)(acc[r] + bias + pe);
    }
  }
  __syncthreads();

  // ---- to_q: (16x128)@(128->128) + bias -> sQB ----
  {
    const int n0 = wave * 16;
    v8f acc = {};
    for (int kc = 0; kc < 128; kc += 32) {
      v16h a = wmma_ld_a(sQA + kc, 128, lane);
      v16h b = wmma_ld_b(wtq + (size_t)n0 * 128 + kc, 128, lane);
      acc = wmma_f16(a, b, acc);
    }
    const int n = n0 + (lane & 15);
    const float bias = btq[n];
    for (int r = 0; r < 8; ++r) {
      int m = r + hi * 8;
      sQB[m * 128 + n] = (_Float16)(acc[r] + bias);
    }
  }
  __syncthreads();

  // ---- k_proj: (64x128)@(128->128) + bias + pos_embed_2d -> sK2 ----
  {
    const int n0 = wave * 16;
    const int n  = n0 + (lane & 15);
    const float bias = bk[n];
    for (int mt = 0; mt < 4; ++mt) {
      v8f acc = {};
      for (int kc = 0; kc < 128; kc += 32) {
        v16h a = wmma_ld_a(sK + mt * 16 * 128 + kc, 128, lane);
        v16h b = wmma_ld_b(wk + (size_t)n0 * 128 + kc, 128, lane);
        acc = wmma_f16(a, b, acc);
      }
      for (int r = 0; r < 8; ++r) {
        int kr = mt * 16 + r + hi * 8;
        float pe = pos_embed_2d[(size_t)t0 * 512 + (size_t)kr * 128 + n];
        sK2[kr * 128 + n] = (_Float16)(acc[r] + bias + pe);
      }
    }
  }
  __syncthreads();

  // ---- to_k: (64x128)@(128->128) + bias -> sK (overwrite key tile) ----
  {
    const int n0 = wave * 16;
    const int n  = n0 + (lane & 15);
    const float bias = btk[n];
    for (int mt = 0; mt < 4; ++mt) {
      v8f acc = {};
      for (int kc = 0; kc < 128; kc += 32) {
        v16h a = wmma_ld_a(sK2 + mt * 16 * 128 + kc, 128, lane);
        v16h b = wmma_ld_b(wtk + (size_t)n0 * 128 + kc, 128, lane);
        acc = wmma_f16(a, b, acc);
      }
      for (int r = 0; r < 8; ++r) {
        int kr = mt * 16 + r + hi * 8;
        sK[kr * 128 + n] = (_Float16)(acc[r] + bias);
      }
    }
  }
  __syncthreads();

  // ---- attention: dot over DH=32, softmax over P=4 ----
  if (tid < 64) {
    int tl = tid >> 2;
    int m  = tid & 3;
    const _Float16* qv = sQB + tl * 128 + m * 32;
    float dots[4];
    float mx = -1e30f;
    for (int p = 0; p < 4; ++p) {
      const _Float16* kv = sK + (tl * 4 + p) * 128 + m * 32;
      float s = 0.f;
      for (int d = 0; d < 32; ++d) s += (float)qv[d] * (float)kv[d];
      s *= 0.17677669529663687f;    // DH^-0.5
      dots[p] = s;
      mx = fmaxf(mx, s);
    }
    float sum = 0.f;
    for (int p = 0; p < 4; ++p) { dots[p] = __expf(dots[p] - mx); sum += dots[p]; }
    float inv = 1.f / sum;
    for (int p = 0; p < 4; ++p) sAtt[tl * 16 + m * 4 + p] = dots[p] * inv;
  }
  __syncthreads();

  // ---- weighted features: x[t, p*128+m*32+dh] = att[t,m,p]*sf[t,p,m,dh] ----
  for (int i = 0; i < 32; ++i) {
    int idx = tid + i * 256;            // 0..8191  (16 tokens * 512 ch)
    int c   = idx & 511;
    int tl  = idx >> 9;
    int p   = c >> 7;
    int m   = (c >> 5) & 3;
    float sf = sampled_feat[(size_t)t0 * 512 + idx];
    xout[(size_t)t0 * 512 + idx] = (_Float16)(sAtt[tl * 16 + m * 4 + p] * sf);
  }
}

// ---------------------------------------------------------------------------
// WMMA GEMM:  out[rows,N] = act( A[rows,K] @ W[N,K]^T + bias )
// 64x128 block tile, 8 waves, 4 output subtiles per wave (A-frag reused 4x).
// K chunked by 32 and DOUBLE-BUFFERED through LDS; the copy uses
// global_load_async_to_lds_b128 + s_wait_asynccnt (inline asm).
// EPI==0 : erf-GELU, f16 output [row*N+col]
// EPI==1 : plain bias, f32 output transposed [col*40000+row]  (final layer)
// ---------------------------------------------------------------------------
template <int K, int EPI>
__global__ __launch_bounds__(256)
void wmma_gemm_kernel(const _Float16* __restrict__ A,
                      const _Float16* __restrict__ W,
                      const float*    __restrict__ bias,
                      void* __restrict__ out, int N)
{
  constexpr int ST = 40;               // padded LDS stride (halves)
  __shared__ _Float16 sA[2][64 * ST];
  __shared__ _Float16 sB[2][128 * ST];

  const int tid  = threadIdx.x;
  const int lane = tid & 31;
  const int wave = tid >> 5;
  const int r0   = blockIdx.x * 64;
  const int n0   = blockIdx.y * 128;

  const int mt  = wave & 3;            // 16-row subtile
  const int ntb = (wave >> 2) * 4;     // four 16-col subtiles

  // A tile 64x32: thread -> (row = tid/4, 8-half slot), one 16B op
  const int liA = tid >> 2, lsA = (tid & 3) * 8;
  // B tile 128x32: thread -> (row = tid/2, 16-half slot), two 16B ops
  const int liB = tid >> 1, lsB = (tid & 1) * 16;

  const _Float16* gA = A + (size_t)(r0 + liA) * K + lsA;
  const _Float16* gB = W + (size_t)(n0 + liB) * K + lsB;

  v8f acc[4] = {{}, {}, {}, {}};

  auto stage = [&](int buf, int kc) {
#if HAVE_ASYNC_LDS
    async_cp16(gA + kc,     &sA[buf][liA * ST + lsA]);
    async_cp16(gB + kc,     &sB[buf][liB * ST + lsB]);
    async_cp16(gB + kc + 8, &sB[buf][liB * ST + lsB + 8]);
#else
    *(v8h*)(&sA[buf][liA * ST + lsA])     = *(const v8h*)(gA + kc);
    *(v8h*)(&sB[buf][liB * ST + lsB])     = *(const v8h*)(gB + kc);
    *(v8h*)(&sB[buf][liB * ST + lsB + 8]) = *(const v8h*)(gB + kc + 8);
#endif
  };

  stage(0, 0);
  int buf = 0;
  for (int kc = 0; kc < K; kc += 32) {
    if (kc + 32 < K) {
      __syncthreads();                 // all waves done reading buf^1
      stage(buf ^ 1, kc + 32);         // async prefetch of next chunk
#if HAVE_ASYNC_LDS
      async_wait3();                   // current chunk's 3 ops have landed
#endif
    } else {
#if HAVE_ASYNC_LDS
      async_wait0();
#endif
    }
    __syncthreads();                   // everyone's data visible

    v16h a = wmma_ld_a(sA[buf] + mt * 16 * ST, ST, lane);
#pragma unroll
    for (int j = 0; j < 4; ++j) {
      v16h b = wmma_ld_b(sB[buf] + (ntb + j) * 16 * ST, ST, lane);
      acc[j] = wmma_f16(a, b, acc[j]);
    }
    buf ^= 1;
  }

  const int nn = lane & 15;
  const int hi = lane >> 4;
#pragma unroll
  for (int j = 0; j < 4; ++j) {
    int col = n0 + (ntb + j) * 16 + nn;
    float bb = bias[col];
#pragma unroll
    for (int r = 0; r < 8; ++r) {
      int row = r0 + mt * 16 + r + hi * 8;
      float v = acc[j][r] + bb;
      if (EPI == 0) {
        ((_Float16*)out)[(size_t)row * N + col] = (_Float16)gelu_erf(v);
      } else {
        ((float*)out)[(size_t)col * Q_TOKENS + row] = v;   // [E,H,W] layout
      }
    }
  }
}

// ---------------------------------------------------------------------------
// Host-side orchestration
// ---------------------------------------------------------------------------
extern "C" void kernel_launch(void* const* d_in, const int* in_sizes, int n_in,
                              void* d_out, int out_size, void* d_ws, size_t ws_size,
                              hipStream_t stream) {
  (void)in_sizes; (void)n_in; (void)out_size; (void)ws_size;

  const float* bev_query     = (const float*)d_in[0];
  const float* sampled_feat  = (const float*)d_in[1];
  const float* bev_pos_embed = (const float*)d_in[2];
  const float* pos_embed_2d  = (const float*)d_in[3];
  const float* q_proj_w = (const float*)d_in[4];
  const float* q_proj_b = (const float*)d_in[5];
  const float* to_q_w   = (const float*)d_in[6];
  const float* to_q_b   = (const float*)d_in[7];
  const float* k_proj_w = (const float*)d_in[8];
  const float* k_proj_b = (const float*)d_in[9];
  const float* to_k_w   = (const float*)d_in[10];
  const float* to_k_b   = (const float*)d_in[11];
  const float* w1 = (const float*)d_in[12];
  const float* b1 = (const float*)d_in[13];
  const float* w2 = (const float*)d_in[14];
  const float* b2 = (const float*)d_in[15];
  const float* w3 = (const float*)d_in[16];
  const float* b3 = (const float*)d_in[17];

  char* ws = (char*)d_ws;
  size_t off = 0;
  auto wsAlloc = [&](size_t bytes) -> void* {
    void* p = ws + off;
    off += (bytes + 255) & ~(size_t)255;
    return p;
  };

  _Float16* wq16  = (_Float16*)wsAlloc((size_t)128 * 256 * 2);
  _Float16* wtq16 = (_Float16*)wsAlloc((size_t)128 * 128 * 2);
  _Float16* wk16  = (_Float16*)wsAlloc((size_t)128 * 128 * 2);
  _Float16* wtk16 = (_Float16*)wsAlloc((size_t)128 * 128 * 2);
  _Float16* w1_16 = (_Float16*)wsAlloc((size_t)1024 * 512 * 2);
  _Float16* w2_16 = (_Float16*)wsAlloc((size_t)1024 * 1024 * 2);
  _Float16* w3_16 = (_Float16*)wsAlloc((size_t)256 * 1024 * 2);
  _Float16* x16   = (_Float16*)wsAlloc((size_t)Q_TOKENS * 512 * 2);
  _Float16* h1    = (_Float16*)wsAlloc((size_t)Q_TOKENS * 1024 * 2);
  _Float16* h2    = (_Float16*)wsAlloc((size_t)Q_TOKENS * 1024 * 2);

  auto cvt = [&](const float* s, _Float16* d, int n) {
    cvt_f32_to_f16_kernel<<<(n + 255) / 256, 256, 0, stream>>>(s, d, n);
  };
  cvt(q_proj_w, wq16,  128 * 256);
  cvt(to_q_w,   wtq16, 128 * 128);
  cvt(k_proj_w, wk16,  128 * 128);
  cvt(to_k_w,   wtk16, 128 * 128);
  cvt(w1, w1_16, 1024 * 512);
  cvt(w2, w2_16, 1024 * 1024);
  cvt(w3, w3_16, 256 * 1024);

  // fused projections + attention  (2500 blocks x 16 tokens)
  frontend_attn_kernel<<<Q_TOKENS / 16, 256, 0, stream>>>(
      bev_query, sampled_feat, bev_pos_embed, pos_embed_2d,
      wq16, q_proj_b, wtq16, to_q_b, wk16, k_proj_b, wtk16, to_k_b, x16);

  // MLP head: 512 -> 1024 (GELU) -> 1024 (GELU) -> 256 (f32, transposed out)
  wmma_gemm_kernel<512, 0><<<dim3(Q_TOKENS / 64, 1024 / 128), 256, 0, stream>>>(
      x16, w1_16, b1, h1, 1024);
  wmma_gemm_kernel<1024, 0><<<dim3(Q_TOKENS / 64, 1024 / 128), 256, 0, stream>>>(
      h1, w2_16, b2, h2, 1024);
  wmma_gemm_kernel<1024, 1><<<dim3(Q_TOKENS / 64, 256 / 128), 256, 0, stream>>>(
      h2, w3_16, b3, d_out, 256);
}
